// ExpertChoiceMoELayer_39779987096112
// MI455X (gfx1250) — compile-verified, matrix operand order checked by reading, not run on hardware
//
#include <hip/hip_runtime.h>
#include <hip/hip_bf16.h>

// Reference shapes: B=4, S=2048, H=1024, E=8, F=2048, T=8192, capacity=1280
#define H_DIM    1024
#define E_NUM    8
#define F_DIM    2048
#define T_NUM    8192
#define CAPACITY 1280   // min(ceil(8192/8*1.25), 8192)

typedef __attribute__((ext_vector_type(16))) __bf16 v16bf;
typedef __attribute__((ext_vector_type(8)))  float  v8f;

union Frag { v16bf v; unsigned u[8]; };

__device__ __forceinline__ unsigned short f2bf(float f) {
  unsigned u = __float_as_uint(f);
  u += 0x7FFFu + ((u >> 16) & 1u);           // round-to-nearest-even
  return (unsigned short)(u >> 16);
}

// ---------------------------------------------------------------------------
// LDS layouts (ISA 7.12.2, wave32, 16-bit operands).
// A tile  : row-major [m][k], stride 32.  Fragment lane = half*16+m, VGPR v
//           holds K-pair at k = (v>>2)*16 + half*8 + (v&3)*2  -> chunk-of-8
//           contiguous ushorts => compiler merges to 2x ds_load_b128.
// B tile  : transposed [n][k], stride 32, with an XOR swizzle on the 16-byte
//           k-chunk keyed by (n>>4)&3 to break the 8-way store conflict
//           (fixed k, n strided by 16) while keeping fragment reads as two
//           aligned b128s (swizzle is uniform across lanes of a fragment).
// ---------------------------------------------------------------------------
__device__ __forceinline__ int bidx(int n, int k) {   // B-tile LDS index
  return n * 32 + ((((k >> 3) ^ ((n >> 4) & 3))) << 3) + (k & 7);
}

__device__ __forceinline__ void load_a_frag(Frag& f, const unsigned short* lA,
                                            int rowbase, int lane) {
  int m = lane & 15, half = lane >> 4;
  const unsigned short* row = lA + (rowbase + m) * 32;
#pragma unroll
  for (int v = 0; v < 8; ++v) {
    int k = ((v >> 2) * 16) + (half * 8) + ((v & 3) * 2);
    f.u[v] = *(const unsigned*)(row + k);
  }
}

__device__ __forceinline__ void load_b_frag(Frag& f, const unsigned short* lBt,
                                            int colbase, int lane) {
  int n = colbase + (lane & 15), half = lane >> 4;
  int sw = (n >> 4) & 3;
  const unsigned short* col = lBt + n * 32;
#pragma unroll
  for (int v = 0; v < 8; ++v) {
    int c = ((v >> 2) * 2) + half;                  // k-chunk 0..3
    f.u[v] = *(const unsigned*)(col + ((c ^ sw) << 3) + ((v & 3) * 2));
  }
}

// ---------------------------------------------------------------------------
// 0) fp32 -> bf16 weight pre-conversion (used when ws_size permits; amortizes
//    the convert across the 20 C-tiles that re-read every weight element).
// ---------------------------------------------------------------------------
__global__ __launch_bounds__(256) void cvt_kernel(
    const float* __restrict__ in, unsigned short* __restrict__ out, int n4) {
  int i = blockIdx.x * 256 + threadIdx.x;
  int stride = gridDim.x * 256;
  for (; i < n4; i += stride) {
    float4 p = ((const float4*)in)[i];
    unsigned lo = (unsigned)f2bf(p.x) | ((unsigned)f2bf(p.y) << 16);
    unsigned hi = (unsigned)f2bf(p.z) | ((unsigned)f2bf(p.w) << 16);
    *(uint2*)(out + (long)i * 4) = make_uint2(lo, hi);
  }
}

// ---------------------------------------------------------------------------
// 1) Gate GEMM: logits[T,E] = x[T,H] @ gate_w[H,E].  Memory-bound, tiny.
// ---------------------------------------------------------------------------
__global__ __launch_bounds__(256) void gate_kernel(
    const float* __restrict__ x, const float* __restrict__ gw,
    float* __restrict__ logits) {
  int tid = blockIdx.x * 256 + threadIdx.x;   // T*E threads
  int t = tid >> 3, e = tid & 7;
  const float* xr = x + (long)t * H_DIM;
  float acc = 0.f;
#pragma unroll 8
  for (int h = 0; h < H_DIM; ++h) acc = fmaf(xr[h], gw[h * E_NUM + e], acc);
  logits[(long)t * E_NUM + e] = acc;
}

// ---------------------------------------------------------------------------
// 2) Per-expert exact top-CAPACITY via MSB-first radix select (4x 8-bit).
// ---------------------------------------------------------------------------
__global__ __launch_bounds__(1024) void topk_kernel(
    const float* __restrict__ logits, int* __restrict__ sel_idx,
    float* __restrict__ sel_scores) {
  int e = blockIdx.x;
  int tid = threadIdx.x;
  __shared__ unsigned hist[256];
  __shared__ unsigned s_prefix, s_remaining, cnt_gt, cnt_eq;

  auto fkey = [](float f) {
    unsigned u = __float_as_uint(f);
    return (u & 0x80000000u) ? ~u : (u | 0x80000000u);  // order-preserving
  };

  if (tid == 0) { s_prefix = 0u; s_remaining = CAPACITY; }

  for (int pass = 3; pass >= 0; --pass) {
    for (int i = tid; i < 256; i += 1024) hist[i] = 0u;
    __syncthreads();
    unsigned prefix = s_prefix;
    int shift = pass * 8;
    for (int i = tid; i < T_NUM; i += 1024) {
      unsigned k = fkey(logits[(long)i * E_NUM + e]);
      if (pass == 3 || (k >> (shift + 8)) == prefix)
        atomicAdd(&hist[(k >> shift) & 255u], 1u);
    }
    __syncthreads();
    if (tid == 0) {
      unsigned cum = 0, rem = s_remaining;
      int d = 255;
      for (; d > 0; --d) {
        if (cum + hist[d] >= rem) break;
        cum += hist[d];
      }
      s_remaining = rem - cum;                     // needed among digit==d
      s_prefix = (prefix << 8) | (unsigned)d;
    }
    __syncthreads();
  }

  unsigned thr = s_prefix;
  unsigned need_eq = s_remaining;
  unsigned G = CAPACITY - need_eq;                 // count strictly greater
  if (tid == 0) { cnt_gt = 0u; cnt_eq = 0u; }
  __syncthreads();

  for (int i = tid; i < T_NUM; i += 1024) {
    float v = logits[(long)i * E_NUM + e];
    unsigned k = fkey(v);
    if (k > thr) {
      unsigned p = atomicAdd(&cnt_gt, 1u);
      sel_idx[(long)e * CAPACITY + p] = i;
      sel_scores[(long)e * CAPACITY + p] = v;
    } else if (k == thr) {
      unsigned p = atomicAdd(&cnt_eq, 1u);
      if (p < need_eq) {
        sel_idx[(long)e * CAPACITY + G + p] = i;
        sel_scores[(long)e * CAPACITY + G + p] = v;
      }
    }
  }
}

// ---------------------------------------------------------------------------
// 3) Per-expert softmax over the CAPACITY selected scores.
// ---------------------------------------------------------------------------
__global__ __launch_bounds__(256) void softmax_kernel(
    const float* __restrict__ sel_scores, float* __restrict__ sel_w) {
  int e = blockIdx.x, tid = threadIdx.x;
  __shared__ float red[256];
  const float* s = sel_scores + (long)e * CAPACITY;
  float m = -3.4e38f;
  for (int i = tid; i < CAPACITY; i += 256) m = fmaxf(m, s[i]);
  red[tid] = m; __syncthreads();
  for (int k = 128; k > 0; k >>= 1) {
    if (tid < k) red[tid] = fmaxf(red[tid], red[tid + k]);
    __syncthreads();
  }
  m = red[0]; __syncthreads();
  float sum = 0.f;
  for (int i = tid; i < CAPACITY; i += 256) sum += __expf(s[i] - m);
  red[tid] = sum; __syncthreads();
  for (int k = 128; k > 0; k >>= 1) {
    if (tid < k) red[tid] += red[tid + k];
    __syncthreads();
  }
  float inv = __builtin_amdgcn_rcpf(red[0]);
  for (int i = tid; i < CAPACITY; i += 256)
    sel_w[(long)e * CAPACITY + i] = __expf(s[i] - m) * inv;
}

// ---------------------------------------------------------------------------
// 4) FFN1: hidden[e,c,f] = silu(Xg@w1) * (Xg@w3), bf16 WMMA, fp32 acc.
// Block = 256 threads (8 waves). WG tile 64(C) x 128(F); wave tile 32x32
// (2x2 WMMA 16x16x32). K over H in steps of 32 through LDS.
// PRE=true: weights already bf16 in workspace (no per-tile convert).
// ---------------------------------------------------------------------------
template <bool PRE>
__global__ __launch_bounds__(256) void ffn1_kernel(
    const float* __restrict__ x,
    const float* __restrict__ w1f, const float* __restrict__ w3f,
    const unsigned short* __restrict__ w1b, const unsigned short* __restrict__ w3b,
    const int* __restrict__ sel_idx, unsigned short* __restrict__ hidden) {
  __shared__ unsigned short lA[64 * 32];
  __shared__ unsigned short lB1[128 * 32];
  __shared__ unsigned short lB3[128 * 32];

  int e = blockIdx.z, c0 = blockIdx.y * 64, f0 = blockIdx.x * 128;
  int tid = threadIdx.x, lane = tid & 31, w = tid >> 5;
  int wm = w >> 2, wn = w & 3;                  // waves: 2(M) x 4(N)

  v8f accA[2][2] = {}, accG[2][2] = {};

  int ar = tid >> 2, aseg = (tid & 3) * 8;      // A: 4 thr/row, 8 floats each
  long tok = sel_idx[(long)e * CAPACITY + c0 + ar];
  const float* aptr = x + tok * H_DIM + aseg;
  int bk = tid >> 3, bseg = (tid & 7) * 16;     // B: 8 thr/K-row, 16 cols each
  long boff = ((long)e * H_DIM + bk) * F_DIM + f0 + bseg;

  for (int kk = 0; kk < H_DIM; kk += 32) {
    {
      float4 p0 = *(const float4*)(aptr + kk);
      float4 p1 = *(const float4*)(aptr + kk + 4);
      unsigned short* d = lA + ar * 32 + aseg;
      d[0] = f2bf(p0.x); d[1] = f2bf(p0.y); d[2] = f2bf(p0.z); d[3] = f2bf(p0.w);
      d[4] = f2bf(p1.x); d[5] = f2bf(p1.y); d[6] = f2bf(p1.z); d[7] = f2bf(p1.w);
    }
    if (PRE) {
      const unsigned* g1 = (const unsigned*)(w1b + boff + (long)kk * F_DIM);
      const unsigned* g3 = (const unsigned*)(w3b + boff + (long)kk * F_DIM);
#pragma unroll
      for (int j = 0; j < 8; ++j) {
        unsigned u1 = g1[j], u3 = g3[j];
        lB1[bidx(bseg + 2 * j,     bk)] = (unsigned short)u1;
        lB1[bidx(bseg + 2 * j + 1, bk)] = (unsigned short)(u1 >> 16);
        lB3[bidx(bseg + 2 * j,     bk)] = (unsigned short)u3;
        lB3[bidx(bseg + 2 * j + 1, bk)] = (unsigned short)(u3 >> 16);
      }
    } else {
      const float* g1 = w1f + boff + (long)kk * F_DIM;
      const float* g3 = w3f + boff + (long)kk * F_DIM;
#pragma unroll
      for (int j = 0; j < 4; ++j) {
        float4 p = *(const float4*)(g1 + j * 4);
        lB1[bidx(bseg + j * 4 + 0, bk)] = f2bf(p.x);
        lB1[bidx(bseg + j * 4 + 1, bk)] = f2bf(p.y);
        lB1[bidx(bseg + j * 4 + 2, bk)] = f2bf(p.z);
        lB1[bidx(bseg + j * 4 + 3, bk)] = f2bf(p.w);
        float4 q = *(const float4*)(g3 + j * 4);
        lB3[bidx(bseg + j * 4 + 0, bk)] = f2bf(q.x);
        lB3[bidx(bseg + j * 4 + 1, bk)] = f2bf(q.y);
        lB3[bidx(bseg + j * 4 + 2, bk)] = f2bf(q.z);
        lB3[bidx(bseg + j * 4 + 3, bk)] = f2bf(q.w);
      }
    }
    __syncthreads();

    Frag a[2], b1f_[2], b3f_[2];
#pragma unroll
    for (int mi = 0; mi < 2; ++mi) load_a_frag(a[mi], lA, wm * 32 + mi * 16, lane);
#pragma unroll
    for (int ni = 0; ni < 2; ++ni) {
      load_b_frag(b1f_[ni], lB1, wn * 32 + ni * 16, lane);
      load_b_frag(b3f_[ni], lB3, wn * 32 + ni * 16, lane);
    }
#pragma unroll
    for (int mi = 0; mi < 2; ++mi)
#pragma unroll
      for (int ni = 0; ni < 2; ++ni) {
        accA[mi][ni] = __builtin_amdgcn_wmma_f32_16x16x32_bf16(
            false, a[mi].v, false, b1f_[ni].v, (short)0, accA[mi][ni], false, false);
        accG[mi][ni] = __builtin_amdgcn_wmma_f32_16x16x32_bf16(
            false, a[mi].v, false, b3f_[ni].v, (short)0, accG[mi][ni], false, false);
      }
    __syncthreads();
  }

  // SwiGLU epilogue -> bf16 hidden; fast v_rcp instead of IEEE divide.
  int n_l = lane & 15, half = lane >> 4;
#pragma unroll
  for (int mi = 0; mi < 2; ++mi)
#pragma unroll
    for (int ni = 0; ni < 2; ++ni) {
      int cbase = c0 + wm * 32 + mi * 16;
      int fcol = f0 + wn * 32 + ni * 16 + n_l;
#pragma unroll
      for (int r = 0; r < 8; ++r) {
        int c = cbase + r + half * 8;
        float av = accA[mi][ni][r], gv = accG[mi][ni][r];
        float hv = av * __builtin_amdgcn_rcpf(1.0f + __expf(-av)) * gv;
        hidden[((long)e * CAPACITY + c) * F_DIM + fcol] = f2bf(hv);
      }
    }
}

// ---------------------------------------------------------------------------
// 5) FFN2: out += weight[e,c] * (hidden @ w2), atomic scatter-add combine.
// A tile (bf16 hidden, row-major, 16B/lane) is staged with the CDNA5 async
// copy engine: global_load_async_to_lds_b128 + s_wait_asynccnt.
// ---------------------------------------------------------------------------
template <bool PRE>
__global__ __launch_bounds__(256) void ffn2_kernel(
    const unsigned short* __restrict__ hidden,
    const float* __restrict__ w2f, const unsigned short* __restrict__ w2b,
    const int* __restrict__ sel_idx, const float* __restrict__ sel_w,
    float* __restrict__ out) {
  __shared__ unsigned short lA[64 * 32];
  __shared__ unsigned short lB[128 * 32];

  int e = blockIdx.z, c0 = blockIdx.y * 64, h0 = blockIdx.x * 128;
  int tid = threadIdx.x, lane = tid & 31, w = tid >> 5;
  int wm = w >> 2, wn = w & 3;

  v8f acc[2][2] = {};

  int ar = tid >> 2, aseg = (tid & 3) * 8;
  // async A staging: 32-bit LDS dest addr (low 32 bits of the flat shared
  // pointer are the LDS offset), 32-bit global byte offset, SGPR base (GVS).
  unsigned lds_dst = (unsigned)(unsigned long long)(lA + ar * 32 + aseg);
  long arow = ((long)e * CAPACITY + c0 + ar) * F_DIM + aseg;   // elements
  int bk = tid >> 3, bseg = (tid & 7) * 16;
  long boff = ((long)e * F_DIM + bk) * H_DIM + h0 + bseg;

  for (int kk = 0; kk < F_DIM; kk += 32) {
    {
      unsigned goff = (unsigned)((arow + kk) * 2);             // bytes
      asm volatile("global_load_async_to_lds_b128 %0, %1, %2"
                   :: "v"(lds_dst), "v"(goff), "s"(hidden) : "memory");
    }
    if (PRE) {
      const unsigned* g = (const unsigned*)(w2b + boff + (long)kk * H_DIM);
#pragma unroll
      for (int j = 0; j < 8; ++j) {
        unsigned u = g[j];
        lB[bidx(bseg + 2 * j,     bk)] = (unsigned short)u;
        lB[bidx(bseg + 2 * j + 1, bk)] = (unsigned short)(u >> 16);
      }
    } else {
      const float* g = w2f + boff + (long)kk * H_DIM;
#pragma unroll
      for (int j = 0; j < 4; ++j) {
        float4 p = *(const float4*)(g + j * 4);
        lB[bidx(bseg + j * 4 + 0, bk)] = f2bf(p.x);
        lB[bidx(bseg + j * 4 + 1, bk)] = f2bf(p.y);
        lB[bidx(bseg + j * 4 + 2, bk)] = f2bf(p.z);
        lB[bidx(bseg + j * 4 + 3, bk)] = f2bf(p.w);
      }
    }
    asm volatile("s_wait_asynccnt 0" ::: "memory");
    __syncthreads();

    Frag a[2], bf_[2];
#pragma unroll
    for (int mi = 0; mi < 2; ++mi) load_a_frag(a[mi], lA, wm * 32 + mi * 16, lane);
#pragma unroll
    for (int ni = 0; ni < 2; ++ni) load_b_frag(bf_[ni], lB, wn * 32 + ni * 16, lane);
#pragma unroll
    for (int mi = 0; mi < 2; ++mi)
#pragma unroll
      for (int ni = 0; ni < 2; ++ni)
        acc[mi][ni] = __builtin_amdgcn_wmma_f32_16x16x32_bf16(
            false, a[mi].v, false, bf_[ni].v, (short)0, acc[mi][ni], false, false);
    __syncthreads();
  }

  int n_l = lane & 15, half = lane >> 4;
#pragma unroll
  for (int mi = 0; mi < 2; ++mi) {
    int cbase = c0 + wm * 32 + mi * 16;
#pragma unroll
    for (int r = 0; r < 8; ++r) {
      int c = cbase + r + half * 8;
      int tok = sel_idx[(long)e * CAPACITY + c];
      float wv = sel_w[(long)e * CAPACITY + c];
      float* orow = out + (long)tok * H_DIM;
#pragma unroll
      for (int ni = 0; ni < 2; ++ni) {
        int hcol = h0 + wn * 32 + ni * 16 + n_l;
        __hip_atomic_fetch_add(&orow[hcol], acc[mi][ni][r] * wv,
                               __ATOMIC_RELAXED, __HIP_MEMORY_SCOPE_AGENT);
      }
    }
  }
}

// ---------------------------------------------------------------------------
extern "C" void kernel_launch(void* const* d_in, const int* in_sizes, int n_in,
                              void* d_out, int out_size, void* d_ws,
                              size_t ws_size, hipStream_t stream) {
  (void)in_sizes; (void)n_in;
  const float* x  = (const float*)d_in[0];
  const float* gw = (const float*)d_in[1];
  const float* w1 = (const float*)d_in[2];
  const float* w2 = (const float*)d_in[3];
  const float* w3 = (const float*)d_in[4];
  float* out = (float*)d_out;   // T*H floats + 1 aux scalar (0.0)

  const size_t WELEMS = (size_t)E_NUM * H_DIM * F_DIM;   // 16M per weight
  char* ws = (char*)d_ws;
  size_t off = 0;
  float* logits = (float*)(ws + off);            off += (size_t)T_NUM * E_NUM * 4;
  int*   sel_idx = (int*)(ws + off);             off += (size_t)E_NUM * CAPACITY * 4;
  float* sel_scores = (float*)(ws + off);        off += (size_t)E_NUM * CAPACITY * 4;
  float* sel_w = (float*)(ws + off);             off += (size_t)E_NUM * CAPACITY * 4;
  unsigned short* hidden = (unsigned short*)(ws + off);
  off += (size_t)E_NUM * CAPACITY * F_DIM * 2;   // 40 MiB
  unsigned short* w1b = (unsigned short*)(ws + off); off += WELEMS * 2;
  unsigned short* w3b = (unsigned short*)(ws + off); off += WELEMS * 2;
  unsigned short* w2b = (unsigned short*)(ws + off); off += WELEMS * 2;
  const bool pre = (ws_size >= off);             // ~136 MB for the fast path

  // 0) zero output (scatter-add target; also the trailing 0.0 scalar)
  hipMemsetAsync(d_out, 0, (size_t)out_size * sizeof(float), stream);

  // 0b) one-time (per call) weight downconvert when workspace allows
  if (pre) {
    int n4 = (int)(WELEMS / 4);
    cvt_kernel<<<2048, 256, 0, stream>>>(w1, w1b, n4);
    cvt_kernel<<<2048, 256, 0, stream>>>(w3, w3b, n4);
    cvt_kernel<<<2048, 256, 0, stream>>>(w2, w2b, n4);
  }

  // 1) gate logits
  gate_kernel<<<(T_NUM * E_NUM) / 256, 256, 0, stream>>>(x, gw, logits);

  // 2) per-expert exact top-CAPACITY
  topk_kernel<<<E_NUM, 1024, 0, stream>>>(logits, sel_idx, sel_scores);

  // 3) per-expert softmax weights
  softmax_kernel<<<E_NUM, 256, 0, stream>>>(sel_scores, sel_w);

  // 4) grouped SwiGLU up-projection (bf16 WMMA)
  dim3 g1(F_DIM / 128, CAPACITY / 64, E_NUM);
  dim3 g2(H_DIM / 128, CAPACITY / 64, E_NUM);
  if (pre) {
    ffn1_kernel<true><<<g1, 256, 0, stream>>>(x, w1, w3, w1b, w3b, sel_idx, hidden);
    ffn2_kernel<true><<<g2, 256, 0, stream>>>(hidden, w2, w2b, sel_idx, sel_w, out);
  } else {
    ffn1_kernel<false><<<g1, 256, 0, stream>>>(x, w1, w3, w1b, w3b, sel_idx, hidden);
    ffn2_kernel<false><<<g2, 256, 0, stream>>>(hidden, w2, w2b, sel_idx, sel_w, out);
  }
}
